// Attention_82403242541756
// MI455X (gfx1250) — matrix operand order repaired and checked by reading, not compiled, not run on hardware
//
#include <hip/hip_runtime.h>
#include <hip/hip_bf16.h>

// ---------------------------------------------------------------------------
// Attention block for MI455X (gfx1250): bf16 WMMA everywhere, f32 softmax.
//   1) cast inputs f32->bf16
//   2) transpose+cast W_qkv, W_out to [N][K] bf16
//   3) QKV GEMM  [6144,1024]x[1024,3072] -> bf16 qkv (+bias)
//   4) transpose V region to [bh][d][t] bf16
//   5) flash attention: S^T = K*Q^T via WMMA; C-layout of S^T == A-operand
//      layout of exp(P) -> in-register softmax->WMMA handoff. K/V tiles are
//      staged to LDS with double-buffered global_load_async_to_lds_b128
//      (ASYNCcnt) and shared by all 8 waves of the workgroup.
//   6) output GEMM [6144,1024]x[1024,1024] -> f32 d_out (+bias)
// ---------------------------------------------------------------------------

typedef __bf16 bf16;
typedef __attribute__((ext_vector_type(16))) __bf16 v16bf;
typedef __attribute__((ext_vector_type(8)))  __bf16 v8bf;
typedef __attribute__((ext_vector_type(8)))  float  v8f;

#define SEQ    2048
#define HDIM   1024
#define HD     64
#define NBH    48          // 3 batches * 16 heads
#define QKV_N  3072
#define MROWS  6144        // 3 * 2048 token rows

static __device__ __forceinline__ bf16 f2bf(float f) {
  unsigned u = __builtin_bit_cast(unsigned, f);
  u += 0x7FFFu + ((u >> 16) & 1u);              // round-to-nearest-even
  unsigned short h = (unsigned short)(u >> 16);
  return __builtin_bit_cast(bf16, h);
}

// Raw v_exp_f32: args are always <= 0 here; deep-negative inputs flush to 0,
// which is exactly the flash-softmax semantic. Avoids libm's guard sequence.
static __device__ __forceinline__ float fast_exp2(float x) {
  return __builtin_amdgcn_exp2f(x);
}

static __device__ __forceinline__ v8f vzero8() {
  v8f z;
#pragma unroll
  for (int i = 0; i < 8; ++i) z[i] = 0.0f;
  return z;
}

// A operand, 16x32 bf16 tile from a row-major [rows][ld] source.
// Lane L holds row M = L%16; lanes 0-15 get K {0..7,16..23}, lanes 16-31 get
// K {8..15,24..31} -> two contiguous 16B loads per lane.
static __device__ __forceinline__ v16bf load_a16x32(const bf16* __restrict__ base, int ld,
                                                    int row_base, int kofs) {
  const int lane = threadIdx.x & 31;
  const bf16* p = base + (size_t)(row_base + (lane & 15)) * ld + kofs + ((lane >> 4) << 3);
  v8bf lo = *(const v8bf*)p;
  v8bf hi = *(const v8bf*)(p + 16);
  return __builtin_shufflevector(lo, hi, 0, 1, 2, 3, 4, 5, 6, 7,
                                         8, 9, 10, 11, 12, 13, 14, 15);
}

// B operand, 32x16 bf16 tile from an N-major ([N][ld-over-K]) source.
// Lane L holds column N = L%16; lanes 0-15 hold K {0..7,16..23}, lanes 16-31
// K {8..15,24..31} -> two contiguous 16B loads per lane.
static __device__ __forceinline__ v16bf load_b32x16(const bf16* __restrict__ base, int ld,
                                                    int col_base, int kofs) {
  const int lane = threadIdx.x & 31;
  const bf16* p = base + (size_t)(col_base + (lane & 15)) * ld + kofs + ((lane >> 4) << 3);
  v8bf lo = *(const v8bf*)p;
  v8bf hi = *(const v8bf*)(p + 16);
  return __builtin_shufflevector(lo, hi, 0, 1, 2, 3, 4, 5, 6, 7,
                                         8, 9, 10, 11, 12, 13, 14, 15);
}

static __device__ __forceinline__ v8f wmma_bf16f32(v16bf a, v16bf b, v8f c) {
  return __builtin_amdgcn_wmma_f32_16x16x32_bf16(false, a, false, b, (short)0, c,
                                                 false, false);
}

// gfx1250 async DMA: global -> LDS, 16 bytes per lane, tracked by ASYNCcnt.
// vdst = LDS byte offset (low 32 bits of the generic LDS address per the
// aperture mapping), vaddr = 64-bit global address, 'off' = no SGPR base.
static __device__ __forceinline__ void async_copy_b128(bf16* lds, const bf16* g) {
  unsigned loff = (unsigned)(unsigned long long)lds;
  asm volatile("global_load_async_to_lds_b128 %0, %1, off"
               :: "v"(loff), "v"(g)
               : "memory");
}

static __device__ __forceinline__ void wait_asynccnt0() {
  asm volatile("s_wait_asynccnt 0x0" ::: "memory");
}

// ---------------------------------------------------------------------------
__global__ __launch_bounds__(256) void cast_f32_bf16(const float* __restrict__ s,
                                                     bf16* __restrict__ d, int n) {
  int i = blockIdx.x * 256 + threadIdx.x;
  if (i < n) d[i] = f2bf(s[i]);
}

// W [K][N] f32  ->  Wt [N][K] bf16 (output-index coalesced)
__global__ __launch_bounds__(256) void transpose_cast(const float* __restrict__ w,
                                                      bf16* __restrict__ wt, int K, int N) {
  int i = blockIdx.x * 256 + threadIdx.x;
  if (i >= K * N) return;
  int n = i / K, k = i - n * K;
  wt[i] = f2bf(w[(size_t)k * N + n]);
}

// V [bh][t][d] bf16 -> Vt [bh][d][t] bf16
__global__ __launch_bounds__(256) void transpose_v(const bf16* __restrict__ v,
                                                   bf16* __restrict__ vt) {
  int i = blockIdx.x * 256 + threadIdx.x;       // over 48*64*2048 = 6291456
  int bh = i >> 17;
  int r  = i & 131071;
  int d  = r >> 11;
  int t  = r & 2047;
  vt[i] = v[((size_t)bh << 17) + (size_t)t * HD + d];
}

// ---------------------------------------------------------------------------
// C[M][N] = A[M][K] * Bt[N][K]^T + bias ; 8 waves/block, 32x32 tile per wave.
template <bool OUT_BF16>
__global__ __launch_bounds__(256) void gemm_bf16_wmma(const bf16* __restrict__ A,
                                                      const bf16* __restrict__ Bt,
                                                      const float* __restrict__ bias,
                                                      void* __restrict__ out,
                                                      int M, int N, int K) {
  const int lane  = threadIdx.x & 31;
  const int wave  = threadIdx.x >> 5;
  const int mbase = blockIdx.y * 64  + (wave & 1) * 32;
  const int nbase = blockIdx.x * 128 + (wave >> 1) * 32;

  v8f acc00 = vzero8(), acc01 = vzero8(), acc10 = vzero8(), acc11 = vzero8();

  for (int k = 0; k < K; k += 32) {
    v16bf a0 = load_a16x32(A, K, mbase, k);
    v16bf a1 = load_a16x32(A, K, mbase + 16, k);
    v16bf b0 = load_b32x16(Bt, K, nbase, k);
    v16bf b1 = load_b32x16(Bt, K, nbase + 16, k);
    acc00 = wmma_bf16f32(a0, b0, acc00);
    acc01 = wmma_bf16f32(a0, b1, acc01);
    acc10 = wmma_bf16f32(a1, b0, acc10);
    acc11 = wmma_bf16f32(a1, b1, acc11);
  }

  const int ncol = lane & 15;
  const int rofs = (lane >> 4) << 3;            // C layout: M = r + 8*(lane/16)
#pragma unroll
  for (int r = 0; r < 8; ++r) {
    int m0 = mbase + rofs + r, m1 = m0 + 16;
    int n0 = nbase + ncol,     n1 = n0 + 16;
    float v00 = acc00[r] + bias[n0];
    float v01 = acc01[r] + bias[n1];
    float v10 = acc10[r] + bias[n0];
    float v11 = acc11[r] + bias[n1];
    if (OUT_BF16) {
      bf16* o = (bf16*)out;
      o[(size_t)m0 * N + n0] = f2bf(v00);
      o[(size_t)m0 * N + n1] = f2bf(v01);
      o[(size_t)m1 * N + n0] = f2bf(v10);
      o[(size_t)m1 * N + n1] = f2bf(v11);
    } else {
      float* o = (float*)out;
      o[(size_t)m0 * N + n0] = v00;
      o[(size_t)m0 * N + n1] = v01;
      o[(size_t)m1 * N + n0] = v10;
      o[(size_t)m1 * N + n1] = v11;
    }
  }
}

// ---------------------------------------------------------------------------
// Stage one 32-key block into LDS: K tile [32][64] + V^T tile [64][32],
// 4 KB each = 256 x b128 async transfers, one K + one V chunk per thread.
static __device__ __forceinline__ void stage_tiles(bf16* __restrict__ dK,
                                                   bf16* __restrict__ dV,
                                                   const bf16* __restrict__ kb,
                                                   const bf16* __restrict__ vtb,
                                                   int t0, int tid) {
  const int kr = tid >> 3, kc = (tid & 7) << 3;   // 32 rows x 8 chunks
  async_copy_b128(dK + kr * HD + kc, kb + (size_t)(t0 + kr) * HD + kc);
  const int vr = tid >> 2, vc = (tid & 3) << 3;   // 64 rows x 4 chunks
  async_copy_b128(dV + vr * 32 + vc, vtb + (size_t)vr * SEQ + t0 + vc);
}

// Flash attention: 8 waves/block, one 16-query tile per wave; all waves share
// double-buffered K/V tiles staged via async LDS DMA.
// S^T = K * Q^T ; C-layout of S^T == A-operand layout of exp(P) for P*V.
__global__ __launch_bounds__(256) void flash_attn_kernel(const bf16* __restrict__ qkv,
                                                         const bf16* __restrict__ vt,
                                                         bf16* __restrict__ ctx) {
  constexpr float SM_L2 = 0.03125f * 1.44269504088896340736f;  // H^-0.5 * log2(e)
  __shared__ bf16 sK[2][32 * HD];   // [t-offset][d]   4 KB per buffer
  __shared__ bf16 sV[2][HD * 32];   // [d][t-offset]   4 KB per buffer

  const int tid   = threadIdx.x;
  const int lane  = tid & 31;
  const int wave  = tid >> 5;                   // 0..7
  const int bh    = blockIdx.x;                 // 0..47
  const int qbase = blockIdx.y * 128 + wave * 16;
  const size_t bho = (size_t)bh * SEQ * HD;

  const bf16* qb  = qkv + bho;                          // Q   [2048][64]
  const bf16* kb  = qkv + (size_t)NBH * SEQ * HD + bho; // K   [2048][64]
  const bf16* vtb = vt + bho;                           // V^T [64][2048]

  // Q tile resident as two B operands (K-dim = d: 0..31 and 32..63)
  v16bf qb0 = load_b32x16(qb, HD, qbase, 0);
  v16bf qb1 = load_b32x16(qb, HD, qbase, 32);

  v8f acc[4];
#pragma unroll
  for (int dt = 0; dt < 4; ++dt) acc[dt] = vzero8();
  float m = -3.0e38f, l = 0.0f;
  const int rofs8 = (lane >> 4) << 3;

  stage_tiles(sK[0], sV[0], kb, vtb, 0, tid);

  for (int tb = 0; tb < SEQ / 32; ++tb) {
    const int cur = tb & 1;
    wait_asynccnt0();        // our async chunks for buffer `cur` have landed
    __syncthreads();         // everyone's chunks landed; prev buffer readers done
    if (tb + 1 < SEQ / 32)
      stage_tiles(sK[cur ^ 1], sV[cur ^ 1], kb, vtb, (tb + 1) * 32, tid);

    const bf16* lk = sK[cur];
    const bf16* lv = sV[cur];
    v16bf ka00 = load_a16x32(lk, HD, 0,  0);
    v16bf ka01 = load_a16x32(lk, HD, 0,  32);
    v16bf ka10 = load_a16x32(lk, HD, 16, 0);
    v16bf ka11 = load_a16x32(lk, HD, 16, 32);

    v8f st0 = vzero8(), st1 = vzero8();
    st0 = wmma_bf16f32(ka00, qb0, st0);
    st0 = wmma_bf16f32(ka01, qb1, st0);
    st1 = wmma_bf16f32(ka10, qb0, st1);
    st1 = wmma_bf16f32(ka11, qb1, st1);

    // Online softmax in log2 domain. Each lane holds 16 key-logits of query
    // s = lane&15; cross-half merge with shfl_xor(16).
    float z0[8], z1[8];
    float mloc = -3.0e38f;
#pragma unroll
    for (int r = 0; r < 8; ++r) {
      z0[r] = st0[r] * SM_L2;
      z1[r] = st1[r] * SM_L2;
      mloc = fmaxf(mloc, fmaxf(z0[r], z1[r]));
    }
    mloc = fmaxf(mloc, __shfl_xor(mloc, 16, 32));
    float mnew  = fmaxf(m, mloc);
    float alpha = fast_exp2(m - mnew);

    float p0[8], p1[8], ls = 0.0f;
#pragma unroll
    for (int r = 0; r < 8; ++r) {
      p0[r] = fast_exp2(z0[r] - mnew);
      p1[r] = fast_exp2(z1[r] - mnew);
      ls += p0[r] + p1[r];
    }
    ls += __shfl_xor(ls, 16, 32);
    l = l * alpha + ls;
    m = mnew;

    // Pack exp(P) straight into a WMMA A operand (layouts coincide).
    v16bf ap;
#pragma unroll
    for (int j = 0; j < 8; ++j) {
      ap[j]     = f2bf(p0[j]);
      ap[j + 8] = f2bf(p1[j]);
    }

    // Rescale running ctx accumulators (row s = r + rofs8 in C layout).
#pragma unroll
    for (int r = 0; r < 8; ++r) {
      float ar = __shfl(alpha, r + rofs8, 32);
#pragma unroll
      for (int dt = 0; dt < 4; ++dt) acc[dt][r] *= ar;
    }

    // ctx += P * V   (V^T rows in LDS are contiguous along t)
#pragma unroll
    for (int dt = 0; dt < 4; ++dt) {
      v16bf vb = load_b32x16(lv, 32, dt * 16, 0);
      acc[dt] = wmma_bf16f32(ap, vb, acc[dt]);
    }
  }

  float rinv = 1.0f / l;
  const int dcol = lane & 15;
#pragma unroll
  for (int r = 0; r < 8; ++r) {
    float sc = __shfl(rinv, r + rofs8, 32);
    int srow = qbase + r + rofs8;
#pragma unroll
    for (int dt = 0; dt < 4; ++dt)
      ctx[bho + (size_t)srow * HD + dt * 16 + dcol] = f2bf(acc[dt][r] * sc);
  }
}

// ---------------------------------------------------------------------------
extern "C" void kernel_launch(void* const* d_in, const int* in_sizes, int n_in,
                              void* d_out, int out_size, void* d_ws, size_t ws_size,
                              hipStream_t stream) {
  (void)in_sizes; (void)n_in; (void)out_size; (void)ws_size;
  const float* x    = (const float*)d_in[0];   // [3,2048,1024]
  const float* wqkv = (const float*)d_in[1];   // [1024,3072]
  const float* bqkv = (const float*)d_in[2];   // [3072]
  const float* wout = (const float*)d_in[3];   // [1024,1024]
  const float* bout = (const float*)d_in[4];   // [1024]

  bf16* Xb  = (bf16*)d_ws;                       // 6,291,456 elems
  bf16* WqT = Xb  + (size_t)MROWS * HDIM;        // 3,145,728
  bf16* WoT = WqT + (size_t)QKV_N * HDIM;        // 1,048,576
  bf16* QKV = WoT + (size_t)HDIM * HDIM;         // 18,874,368
  bf16* Vt  = QKV + (size_t)MROWS * QKV_N;       // 6,291,456
  bf16* CTX = Vt  + (size_t)NBH * HD * SEQ;      // 6,291,456  (~84 MB total)

  cast_f32_bf16<<<(MROWS * HDIM) / 256, 256, 0, stream>>>(x, Xb, MROWS * HDIM);
  transpose_cast<<<(QKV_N * HDIM) / 256, 256, 0, stream>>>(wqkv, WqT, HDIM, QKV_N);
  transpose_cast<<<(HDIM * HDIM) / 256, 256, 0, stream>>>(wout, WoT, HDIM, HDIM);

  gemm_bf16_wmma<true><<<dim3(QKV_N / 128, MROWS / 64), 256, 0, stream>>>(
      Xb, WqT, bqkv, QKV, MROWS, QKV_N, HDIM);

  transpose_v<<<(NBH * HD * SEQ) / 256, 256, 0, stream>>>(
      QKV + (size_t)2 * SEQ * QKV_N, Vt);

  flash_attn_kernel<<<dim3(NBH, SEQ / 128), 256, 0, stream>>>(QKV, Vt, CTX);

  gemm_bf16_wmma<false><<<dim3(HDIM / 128, MROWS / 64), 256, 0, stream>>>(
      CTX, WoT, bout, d_out, MROWS, HDIM, HDIM);
}
